// GNN_89816356094415
// MI455X (gfx1250) — compile-verified, matrix operand order checked by reading, not run on hardware
//
#include <hip/hip_runtime.h>

// ---------- types ----------
typedef __attribute__((ext_vector_type(16))) __bf16 v16bf;
typedef __attribute__((ext_vector_type(8)))  __bf16 v8bf;
typedef __attribute__((ext_vector_type(8)))  float  v8f;
typedef __attribute__((ext_vector_type(4)))  float  v4f;

#define HD 256   // hidden dim (H) — hardcoded for shift arithmetic

// ---------- small utility kernels ----------
__global__ void zero_kernel(float* p, int n) {
    int i = blockIdx.x * blockDim.x + threadIdx.x;
    if (i < n) p[i] = 0.0f;
}

__global__ void deg_init_kernel(float* deg, int n) {
    int i = blockIdx.x * blockDim.x + threadIdx.x;
    if (i < n) deg[i] = 1.0f;          // self-loop weight
}

__global__ void deg_acc_kernel(const int* __restrict__ dst, const float* __restrict__ ew,
                               float* deg, int e) {
    int i = blockIdx.x * blockDim.x + threadIdx.x;
    if (i < e) atomicAdd(&deg[dst[i]], ew[i]);
}

__global__ void dis_kernel(const float* __restrict__ deg, float* dis, int n) {
    int i = blockIdx.x * blockDim.x + threadIdx.x;
    if (i < n) dis[i] = rsqrtf(deg[i]);
}

__global__ void norm_kernel(const int* __restrict__ src, const int* __restrict__ dst,
                            const float* __restrict__ ew, const float* __restrict__ dis,
                            float* norm, int e) {
    int i = blockIdx.x * blockDim.x + threadIdx.x;
    if (i < e) norm[i] = dis[src[i]] * ew[i] * dis[dst[i]];
}

// W[k*Nc+n] (fp32, row-major K x Nc) -> WT[n*K+k] (bf16, row-major Nc x K)
__global__ void wtr_kernel(const float* __restrict__ W, __bf16* __restrict__ WT,
                           int K, int Nc) {
    int t = blockIdx.x * blockDim.x + threadIdx.x;
    if (t >= K * Nc) return;
    int k = t / Nc, n = t % Nc;
    WT[(size_t)n * K + k] = (__bf16)W[t];
}

// ---------- WMMA GEMM: C[M,256] = A[M,K] (fp32) x BT[256,K]^T (bf16) ----------
// one wave computes a 16(M) x 64(N) strip: 4 accumulators share one A fragment,
// so each expensive fp32 A load + bf16 conversion feeds 4 v_wmma_f32_16x16x32_bf16.
__global__ __launch_bounds__(256)
void gemm_wmma_kernel(const float* __restrict__ A, const __bf16* __restrict__ BT,
                      float* __restrict__ C, int M, int K) {
    const int lane = threadIdx.x & 31;
    const int wave = threadIdx.x >> 5;
    const int stripsN = HD / 64;                       // 4 strips across N
    const int strip = blockIdx.x * 8 + wave;
    const int stripsTotal = (M >> 4) * stripsN;
    if (strip >= stripsTotal) return;                  // wave-uniform exit (EXEC all-1 for WMMA)
    const int tm  = strip / stripsN;
    const int tn0 = (strip - tm * stripsN) * 4;        // first of 4 consecutive 16-col tiles

    // CDNA5 16-bit A layout: lanes 0-15 -> K = {k0..k0+7, k0+16..k0+23}
    //                        lanes16-31 -> K = {k0+8..k0+15, k0+24..k0+31}
    const int mrow = tm * 16 + (lane & 15);
    const int koff = (lane < 16) ? 0 : 8;
    const float*  arow = A  + (size_t)mrow * K + koff;
    const __bf16* brow = BT + (size_t)(tn0 * 16 + (lane & 15)) * K + koff;
    const size_t  bstride = (size_t)16 * K;            // next 16-col tile of weights

    v8f acc0 = {}, acc1 = {}, acc2 = {}, acc3 = {};
    for (int k0 = 0; k0 < K; k0 += 32) {
        // ---- A fragment: fp32 loads, convert to bf16 in-register ----
        v4f a0 = *(const v4f*)(arow + k0);
        v4f a1 = *(const v4f*)(arow + k0 + 4);
        v4f a2 = *(const v4f*)(arow + k0 + 16);
        v4f a3 = *(const v4f*)(arow + k0 + 20);
        v16bf av;
        #pragma unroll
        for (int i = 0; i < 4; ++i) {
            av[i]      = (__bf16)a0[i];
            av[4 + i]  = (__bf16)a1[i];
            av[8 + i]  = (__bf16)a2[i];
            av[12 + i] = (__bf16)a3[i];
        }
        // ---- 4 B fragments (bf16 weights, cache-resident), 4 WMMAs ----
        #pragma unroll
        for (int j = 0; j < 4; ++j) {
            const __bf16* bp = brow + (size_t)j * bstride + k0;
            v8bf b0 = *(const v8bf*)bp;
            v8bf b1 = *(const v8bf*)(bp + 16);
            v16bf bv;
            #pragma unroll
            for (int i = 0; i < 8; ++i) { bv[i] = b0[i]; bv[8 + i] = b1[i]; }
            v8f* accp = (j == 0) ? &acc0 : (j == 1) ? &acc1 : (j == 2) ? &acc2 : &acc3;
            *accp = __builtin_amdgcn_wmma_f32_16x16x32_bf16(
                /*neg_a=*/false, av, /*neg_b=*/false, bv,
                /*c_mod=*/(short)0, *accp, /*reuse_a=*/false, /*reuse_b=*/false);
        }
    }

    // C layout: VGPR r -> row r (lanes 0-15) / row 8+r (lanes 16-31); col = lane&15
    const int row0 = tm * 16 + ((lane >> 4) ? 8 : 0);
    const int colb = lane & 15;
    #pragma unroll
    for (int j = 0; j < 4; ++j) {
        const v8f& acc = (j == 0) ? acc0 : (j == 1) ? acc1 : (j == 2) ? acc2 : acc3;
        const int col = (tn0 + j) * 16 + colb;
        #pragma unroll
        for (int r = 0; r < 8; ++r)
            C[(size_t)(row0 + r) * HD + col] = acc[r];
    }
}

// ---------- aggregation ----------
// self-loop contribution: agg = g * dis^2  (i indexes N*H)
__global__ void agg_init_kernel(const float* __restrict__ g, const float* __restrict__ dis,
                                float* agg, int total) {
    int i = blockIdx.x * blockDim.x + threadIdx.x;
    if (i >= total) return;
    float d = dis[i >> 8];
    agg[i] = g[i] * d * d;
}

// edge scatter: thread = (edge, 4-feature chunk); 4 fp32 global atomics (land in L2; agg fits L2)
__global__ void agg_edges_kernel(const int* __restrict__ src, const int* __restrict__ dst,
                                 const float* __restrict__ norm, const float* __restrict__ g,
                                 float* agg, int e) {
    size_t t = (size_t)blockIdx.x * blockDim.x + threadIdx.x;
    int ei = (int)(t >> 6);                       // H/4 = 64 chunks per edge
    if (ei >= e) return;
    int c = (int)(t & 63) << 2;
    int s = src[ei], d = dst[ei];
    float w = norm[ei];
    v4f hv = *(const v4f*)(g + (size_t)s * HD + c);
    float* ap = agg + (size_t)d * HD + c;
    atomicAdd(ap + 0, w * hv[0]);
    atomicAdd(ap + 1, w * hv[1]);
    atomicAdd(ap + 2, w * hv[2]);
    atomicAdd(ap + 3, w * hv[3]);
}

__global__ void finalize_kernel(const float* __restrict__ agg, const float* __restrict__ bias,
                                float* h, int total, int do_relu) {
    int i = blockIdx.x * blockDim.x + threadIdx.x;
    if (i >= total) return;
    float v = agg[i] + bias[i & (HD - 1)];
    h[i] = do_relu ? fmaxf(v, 0.0f) : v;
}

// ---------- pooling (batch is sorted) + head ----------
__global__ __launch_bounds__(256)
void pool_kernel(const float* __restrict__ h, const int* __restrict__ batch,
                 float* sums, float* cnt, int n) {
    int f = threadIdx.x;                 // feature 0..255
    int n0 = blockIdx.x * 256;
    if (n0 >= n) return;
    int n1 = min(n0 + 256, n);
    float acc = 0.0f, cc = 0.0f;
    int cur = batch[n0];
    for (int i = n0; i < n1; ++i) {
        int bi = batch[i];
        if (bi != cur) {
            atomicAdd(&sums[cur * HD + f], acc); acc = 0.0f;
            if (f == 0) { atomicAdd(&cnt[cur], cc); cc = 0.0f; }
            cur = bi;
        }
        acc += h[(size_t)i * HD + f];
        if (f == 0) cc += 1.0f;
    }
    atomicAdd(&sums[cur * HD + f], acc);
    if (f == 0) atomicAdd(&cnt[cur], cc);
}

__global__ void head_kernel(const float* __restrict__ sums, const float* __restrict__ cnt,
                            const float* __restrict__ Wlin, const float* __restrict__ blin,
                            float* out) {
    int t = threadIdx.x;                  // 16 outputs: b = t/2, c = t%2
    if (t >= 16) return;
    int b = t >> 1, c = t & 1;
    float inv = 1.0f / fmaxf(cnt[b], 1.0f);
    float acc = 0.0f;
    for (int f = 0; f < HD; ++f)
        acc += sums[b * HD + f] * inv * Wlin[f * 2 + c];
    out[t] = acc + blin[c];
}

// ---------- launch ----------
extern "C" void kernel_launch(void* const* d_in, const int* in_sizes, int n_in,
                              void* d_out, int out_size, void* d_ws, size_t ws_size,
                              hipStream_t stream) {
    const int FIN = 1152;
    const int Nn = in_sizes[0] / FIN;      // 50000
    const int Ee = in_sizes[1] / 2;        // 800000

    const float* x    = (const float*)d_in[0];
    const int*   eidx = (const int*)d_in[1];
    const int*   src  = eidx;
    const int*   dst  = eidx + Ee;
    const float* ew   = (const float*)d_in[2];
    const int*   batch= (const int*)d_in[3];
    const float* W1   = (const float*)d_in[4];
    const float* b1   = (const float*)d_in[5];
    const float* W2   = (const float*)d_in[6];
    const float* b2   = (const float*)d_in[7];
    const float* W3   = (const float*)d_in[8];
    const float* b3   = (const float*)d_in[9];
    const float* Wlin = (const float*)d_in[10];
    const float* blin = (const float*)d_in[11];
    float* out = (float*)d_out;

    // workspace carve-out (256B aligned)
    char* ws = (char*)d_ws;
    size_t off = 0;
    auto alloc = [&](size_t bytes) -> void* {
        off = (off + 255) & ~(size_t)255;
        void* p = ws + off;
        off += bytes;
        return p;
    };
    float*  deg  = (float*)alloc((size_t)Nn * 4);
    float*  dis  = (float*)alloc((size_t)Nn * 4);
    float*  nrm  = (float*)alloc((size_t)Ee * 4);
    __bf16* W1T  = (__bf16*)alloc((size_t)FIN * HD * 2);
    __bf16* W2T  = (__bf16*)alloc((size_t)HD * HD * 2);
    __bf16* W3T  = (__bf16*)alloc((size_t)HD * HD * 2);
    float*  g    = (float*)alloc((size_t)Nn * HD * 4);
    float*  agg  = (float*)alloc((size_t)Nn * HD * 4);
    float*  h    = (float*)alloc((size_t)Nn * HD * 4);
    float*  sums = (float*)alloc((size_t)(8 * HD + 8) * 4);  // sums[8][256] then cnt[8], contiguous
    float*  cnt  = sums + 8 * HD;

    const int NH = Nn * HD;                         // 12.8M
    const int blkNH = (NH + 255) / 256;
    const int blkN  = (Nn + 255) / 256;
    const int blkE  = (Ee + 255) / 256;
    const int strips = (Nn >> 4) * (HD / 64);                        // 16M x 64N strips
    const int gemmBlocks = (strips + 7) / 8;                         // 8 waves/block
    const size_t edgeThreads = (size_t)Ee * (HD / 4);
    const int blkEdge = (int)((edgeThreads + 255) / 256);

    // weights -> bf16, transposed [Nc][K]
    wtr_kernel<<<(FIN * HD + 255) / 256, 256, 0, stream>>>(W1, W1T, FIN, HD);
    wtr_kernel<<<(HD * HD + 255) / 256, 256, 0, stream>>>(W2, W2T, HD, HD);
    wtr_kernel<<<(HD * HD + 255) / 256, 256, 0, stream>>>(W3, W3T, HD, HD);

    // degree / rsqrt / edge norms (computed once, reused for all 3 layers)
    deg_init_kernel<<<blkN, 256, 0, stream>>>(deg, Nn);
    deg_acc_kernel<<<blkE, 256, 0, stream>>>(dst, ew, deg, Ee);
    dis_kernel<<<blkN, 256, 0, stream>>>(deg, dis, Nn);
    norm_kernel<<<blkE, 256, 0, stream>>>(src, dst, ew, dis, nrm, Ee);

    // ---- layer 1 ----
    gemm_wmma_kernel<<<gemmBlocks, 256, 0, stream>>>(x, W1T, g, Nn, FIN);
    agg_init_kernel<<<blkNH, 256, 0, stream>>>(g, dis, agg, NH);
    agg_edges_kernel<<<blkEdge, 256, 0, stream>>>(src, dst, nrm, g, agg, Ee);
    finalize_kernel<<<blkNH, 256, 0, stream>>>(agg, b1, h, NH, 1);

    // ---- layer 2 ----
    gemm_wmma_kernel<<<gemmBlocks, 256, 0, stream>>>(h, W2T, g, Nn, HD);
    agg_init_kernel<<<blkNH, 256, 0, stream>>>(g, dis, agg, NH);
    agg_edges_kernel<<<blkEdge, 256, 0, stream>>>(src, dst, nrm, g, agg, Ee);
    finalize_kernel<<<blkNH, 256, 0, stream>>>(agg, b2, h, NH, 1);

    // ---- layer 3 (no relu) ----
    gemm_wmma_kernel<<<gemmBlocks, 256, 0, stream>>>(h, W3T, g, Nn, HD);
    agg_init_kernel<<<blkNH, 256, 0, stream>>>(g, dis, agg, NH);
    agg_edges_kernel<<<blkEdge, 256, 0, stream>>>(src, dst, nrm, g, agg, Ee);
    finalize_kernel<<<blkNH, 256, 0, stream>>>(agg, b3, h, NH, 0);

    // ---- pooling + head ----
    zero_kernel<<<(8 * HD + 8 + 255) / 256, 256, 0, stream>>>(sums, 8 * HD + 8);
    pool_kernel<<<(Nn + 255) / 256, 256, 0, stream>>>(h, batch, sums, cnt, Nn);
    head_kernel<<<1, 32, 0, stream>>>(sums, cnt, Wlin, blin, out);
}
